// AVWDCRNN_29970281792127
// MI455X (gfx1250) — compile-verified, matrix operand order checked by reading, not run on hardware
//
#include <hip/hip_runtime.h>
#include <hip/hip_bf16.h>
#include <math.h>

typedef _Float16 f16;
typedef f16   v16h  __attribute__((ext_vector_type(16)));
typedef f16   half8 __attribute__((ext_vector_type(8)));
typedef float v8f   __attribute__((ext_vector_type(8)));

#define NN   1024
#define BB   8
#define TT   12
#define DO   64
#define ORD  8
#define EDIM 10

#define WMMA16(a, b, c) \
  __builtin_amdgcn_wmma_f32_16x16x32_f16(false, (a), false, (b), (short)0, (c), false, false)

// ---------------- WMMA fragment helper (CDNA5 16x16x32 f16 layout) ----------------
// lane&15 picks row (A) / column (transposed-B); two contiguous 16-byte half8
// chunks at K = kb and K = 16+kb, kb = 8*(lane>=16)  ->  two global_load_b128.
__device__ __forceinline__ v16h load_frag16(const f16* __restrict__ base, int ld, int lane) {
  const int  mj = lane & 15;
  const int  kb = (lane >> 4) << 3;            // 0 or 8
  const f16* p  = base + (size_t)mj * ld + kb;
  union { v16h v; half8 h[2]; } u;
  u.h[0] = *(const half8*)(p);
  u.h[1] = *(const half8*)(p + 16);
  return u.v;
}

// ---------------- Phase 0 kernels ----------------
// S = softmax(relu(E E^T), axis=1), stored f16 [N,N] (row-major: A operand)
__global__ void __launch_bounds__(256) k_compute_S(const float* __restrict__ E,
                                                   f16* __restrict__ S16) {
  const int n   = blockIdx.x;
  const int tid = threadIdx.x;
  __shared__ float red[256];
  float En[EDIM];
#pragma unroll
  for (int e = 0; e < EDIM; ++e) En[e] = E[n * EDIM + e];
  float vals[4];
  float mx = -1e30f;
#pragma unroll
  for (int j = 0; j < 4; ++j) {
    const int m = tid + j * 256;
    float d = 0.f;
#pragma unroll
    for (int e = 0; e < EDIM; ++e) d += En[e] * E[m * EDIM + e];
    d = fmaxf(d, 0.f);                                  // relu
    vals[j] = d;
    mx = fmaxf(mx, d);
  }
  red[tid] = mx; __syncthreads();
  for (int s = 128; s > 0; s >>= 1) { if (tid < s) red[tid] = fmaxf(red[tid], red[tid + s]); __syncthreads(); }
  mx = red[0]; __syncthreads();
  float sum = 0.f;
#pragma unroll
  for (int j = 0; j < 4; ++j) { vals[j] = __expf(vals[j] - mx); sum += vals[j]; }
  red[tid] = sum; __syncthreads();
  for (int s = 128; s > 0; s >>= 1) { if (tid < s) red[tid] += red[tid + s]; __syncthreads(); }
  const float inv = 1.f / red[0];
#pragma unroll
  for (int j = 0; j < 4; ++j)
    S16[(size_t)n * NN + tid + j * 256] = (f16)(vals[j] * inv);
}

// pool f32 [ED,K=2,Cin,Cout] -> TRANSPOSED f16 [ED, Cout, KC=2*CP]
__global__ void k_pack_pool_t(const float* __restrict__ src, f16* __restrict__ dst,
                              int Cin, int CP, int Cout) {
  const int KC = 2 * CP;
  const int total = EDIM * KC * Cout;
  int idx = blockIdx.x * blockDim.x + threadIdx.x;
  if (idx >= total) return;
  const int o   = idx % Cout;
  const int rem = idx / Cout;
  const int j   = rem % KC;
  const int e   = rem / KC;
  const int k = j / CP, i = j % CP;
  float v = 0.f;
  if (i < Cin) v = src[(((size_t)e * 2 + k) * Cin + i) * Cout + o];
  dst[((size_t)e * Cout + o) * KC + j] = (f16)v;
}

// bias[n,o] = sum_e E[n,e] * gb[e,o]
__global__ void k_mix_bias(const float* __restrict__ E, const float* __restrict__ gb,
                           float* __restrict__ out, int Cout) {
  int idx = blockIdx.x * blockDim.x + threadIdx.x;
  if (idx >= NN * Cout) return;
  const int o = idx % Cout, n = idx / Cout;
  float s = 0.f;
#pragma unroll
  for (int e = 0; e < EDIM; ++e) s += E[n * EDIM + e] * gb[e * Cout + o];
  out[idx] = s;
}

// ---------------- Per-timestep kernels ----------------
// Build concat(x_t, state, c_read):
//  - row-major into Xflat[bn, 0:CP]  (A operand of the node-pool GEMMs)
//  - transposed into Xt[b, i, n]     (B operand of the S-spmm)
__global__ void __launch_bounds__(256) k_build_xcat(int layer, int t,
    const float* __restrict__ xsrc, const float* __restrict__ state,
    const float* __restrict__ cbuf, const float* __restrict__ readv,
    f16* __restrict__ X, f16* __restrict__ Xt, int CP, int Din) {
  int idx = blockIdx.x * blockDim.x + threadIdx.x;
  if (idx >= BB * NN * DO) return;
  const int d  = idx & (DO - 1);
  const int bn = idx >> 6;
  const int n  = bn & (NN - 1);
  const int b  = bn >> 10;
  f16* row = X + (size_t)bn * (2 * CP);
  f16* Xtb = Xt + (size_t)b * CP * NN;
  float cr = 0.f;
  const float* cp = cbuf + ((size_t)bn * DO + d) * ORD;
#pragma unroll
  for (int o = 0; o < ORD; ++o) cr += cp[o] * readv[o];
  const f16 hs = (f16)state[idx];
  const f16 hr = (f16)cr;
  row[Din + d]      = hs;  Xtb[(size_t)(Din + d) * NN + n]      = hs;
  row[Din + DO + d] = hr;  Xtb[(size_t)(Din + DO + d) * NN + n] = hr;
  if (layer == 0) {
    if (d < 2) {
      const f16 hx = (f16)xsrc[(((size_t)b * TT + t) * NN + n) * 2 + d];   // x [B,T,N,2]
      row[d] = hx;  Xtb[(size_t)d * NN + n] = hx;
    }
    if (d >= 34) {                                                          // pad cols 130..159
      row[130 + (d - 34)] = (f16)0.f;
      Xtb[(size_t)(130 + (d - 34)) * NN + n] = (f16)0.f;
    }
  } else {
    const f16 hx = (f16)xsrc[(((size_t)t * BB + b) * NN + n) * DO + d];     // hs0 [T,B,N,64]
    row[d] = hx;  Xtb[(size_t)d * NN + n] = hx;
  }
}

// slot1[n, :] = sum_m S[n,m] * slot0[m, :]
// 2x2 register-blocked: each wave produces a 32x32 output block -> 8 b128 loads
// feed 4 WMMAs per k-step.
__global__ void __launch_bounds__(128) k_spmm(const f16* __restrict__ S16,
                                              const f16* __restrict__ Xt,
                                              f16* __restrict__ X, int CP, int ntiles2) {
  const int lane = threadIdx.x & 31;
  const int wave = threadIdx.x >> 5;
  const int nt2 = blockIdx.y * 4 + wave;
  if (nt2 >= ntiles2) return;               // whole-wave exit: EXEC stays all-ones
  const int mt2 = blockIdx.x;               // 0..31 (rows in 32-row blocks)
  const int b   = blockIdx.z;
  const int ld  = 2 * CP;
  const f16* A0  = S16 + (size_t)mt2 * 32 * NN;
  const f16* A1  = A0 + (size_t)16 * NN;
  const f16* Bt0 = Xt + ((size_t)b * CP + nt2 * 32) * NN;
  const f16* Bt1 = Bt0 + (size_t)16 * NN;
  v8f acc00 = {}, acc01 = {}, acc10 = {}, acc11 = {};
  for (int ks = 0; ks < NN / 32; ++ks) {
    __builtin_prefetch(A0 + (size_t)(ks + 1) * 32, 0, 1);   // global_prefetch_b8
    v16h a0 = load_frag16(A0 + ks * 32, NN, lane);
    v16h a1 = load_frag16(A1 + ks * 32, NN, lane);
    v16h b0 = load_frag16(Bt0 + ks * 32, NN, lane);
    v16h b1 = load_frag16(Bt1 + ks * 32, NN, lane);
    acc00 = WMMA16(a0, b0, acc00);
    acc01 = WMMA16(a0, b1, acc01);
    acc10 = WMMA16(a1, b0, acc10);
    acc11 = WMMA16(a1, b1, acc11);
  }
  f16* out = X + ((size_t)b * NN + (size_t)mt2 * 32) * ld + CP + nt2 * 32;
  const int col = lane & 15, rb = (lane >> 4) << 3;
#pragma unroll
  for (int v = 0; v < 8; ++v) {
    out[(size_t)(rb + v) * ld + col]           = (f16)acc00[v];
    out[(size_t)(rb + v) * ld + 16 + col]      = (f16)acc01[v];
    out[(size_t)(rb + v + 16) * ld + col]      = (f16)acc10[v];
    out[(size_t)(rb + v + 16) * ld + 16 + col] = (f16)acc11[v];
  }
}

// 2x2-blocked: 32 rows x 32 cols per wave.
// acc = sum_e E[n,e] * (Xflat[m-block, :KC] @ W_e[:KC, o-block]); Wt is [e][Cout][KC]
__device__ __forceinline__ void nodepool_acc2(const f16* __restrict__ A0, int KC,
                                              const f16* __restrict__ Wt, int Cout,
                                              const float* __restrict__ E,
                                              int m0, int o0, int lane,
                                              v8f& acc00, v8f& acc01,
                                              v8f& acc10, v8f& acc11) {
  const int ksteps = KC / 32;
  const int rb = (lane >> 4) << 3;
  const f16* A1 = A0 + (size_t)16 * KC;
  for (int e = 0; e < EDIM; ++e) {
    v8f p00 = {}, p01 = {}, p10 = {}, p11 = {};
    const f16* We0 = Wt + ((size_t)e * Cout + o0) * KC;
    const f16* We1 = We0 + (size_t)16 * KC;
    for (int ks = 0; ks < ksteps; ++ks) {
      v16h a0 = load_frag16(A0 + ks * 32, KC, lane);
      v16h a1 = load_frag16(A1 + ks * 32, KC, lane);
      v16h b0 = load_frag16(We0 + ks * 32, KC, lane);
      v16h b1 = load_frag16(We1 + ks * 32, KC, lane);
      p00 = WMMA16(a0, b0, p00);
      p01 = WMMA16(a0, b1, p01);
      p10 = WMMA16(a1, b0, p10);
      p11 = WMMA16(a1, b1, p11);
    }
#pragma unroll
    for (int v = 0; v < 8; ++v) {
      const float s0 = E[((m0 + rb + v) & (NN - 1)) * EDIM + e];
      const float s1 = E[((m0 + 16 + rb + v) & (NN - 1)) * EDIM + e];
      acc00[v] += s0 * p00[v];  acc01[v] += s0 * p01[v];
      acc10[v] += s1 * p10[v];  acc11[v] += s1 * p11[v];
    }
  }
}

__global__ void __launch_bounds__(128) k_gemm_gate(const f16* __restrict__ X, int KC,
    const f16* __restrict__ Wt, const float* __restrict__ bias,
    const float* __restrict__ E, float* __restrict__ zr) {
  const int lane = threadIdx.x & 31;
  const int wave = threadIdx.x >> 5;
  const int m0 = blockIdx.x * 32;
  const int o0 = wave * 32;                 // 4 waves cover 128 output cols
  v8f acc00 = {}, acc01 = {}, acc10 = {}, acc11 = {};
  nodepool_acc2(X + (size_t)m0 * KC, KC, Wt, 2 * DO, E, m0, o0, lane,
                acc00, acc01, acc10, acc11);
  const int col = lane & 15, rb = (lane >> 4) << 3;
#pragma unroll
  for (int v = 0; v < 8; ++v) {
    const int r0 = m0 + rb + v, r1 = r0 + 16;
    const int n0 = r0 & (NN - 1), n1 = r1 & (NN - 1);
    float y;
    y = acc00[v] + bias[n0 * 128 + o0 + col];       zr[(size_t)r0 * 128 + o0 + col]      = 1.f / (1.f + __expf(-y));
    y = acc01[v] + bias[n0 * 128 + o0 + 16 + col];  zr[(size_t)r0 * 128 + o0 + 16 + col] = 1.f / (1.f + __expf(-y));
    y = acc10[v] + bias[n1 * 128 + o0 + col];       zr[(size_t)r1 * 128 + o0 + col]      = 1.f / (1.f + __expf(-y));
    y = acc11[v] + bias[n1 * 128 + o0 + 16 + col];  zr[(size_t)r1 * 128 + o0 + 16 + col] = 1.f / (1.f + __expf(-y));
  }
}

__global__ void __launch_bounds__(128) k_gemm_update(const f16* __restrict__ X, int KC,
    const f16* __restrict__ Wt, const float* __restrict__ bias,
    const float* __restrict__ E, float* __restrict__ hc) {
  const int lane = threadIdx.x & 31;
  const int wave = threadIdx.x >> 5;
  const int o0 = wave * 32;
  if (o0 >= DO) return;                     // waves 2,3 idle (whole-wave exit)
  const int m0 = blockIdx.x * 32;
  v8f acc00 = {}, acc01 = {}, acc10 = {}, acc11 = {};
  nodepool_acc2(X + (size_t)m0 * KC, KC, Wt, DO, E, m0, o0, lane,
                acc00, acc01, acc10, acc11);
  const int col = lane & 15, rb = (lane >> 4) << 3;
#pragma unroll
  for (int v = 0; v < 8; ++v) {
    const int r0 = m0 + rb + v, r1 = r0 + 16;
    const int n0 = r0 & (NN - 1), n1 = r1 & (NN - 1);
    hc[(size_t)r0 * DO + o0 + col]      = tanhf(acc00[v] + bias[n0 * DO + o0 + col]);
    hc[(size_t)r0 * DO + o0 + 16 + col] = tanhf(acc01[v] + bias[n0 * DO + o0 + 16 + col]);
    hc[(size_t)r1 * DO + o0 + col]      = tanhf(acc10[v] + bias[n1 * DO + o0 + col]);
    hc[(size_t)r1 * DO + o0 + 16 + col] = tanhf(acc11[v] + bias[n1 * DO + o0 + 16 + col]);
  }
}

// slot0 state-cols <- z * state (both row-major and transposed copies)
__global__ void __launch_bounds__(256) k_patch_xu(const float* __restrict__ zr,
    const float* __restrict__ state, f16* __restrict__ X, f16* __restrict__ Xt,
    int CP, int Din) {
  int idx = blockIdx.x * blockDim.x + threadIdx.x;
  if (idx >= BB * NN * DO) return;
  const int d  = idx & (DO - 1);
  const int bn = idx >> 6;
  const int n  = bn & (NN - 1);
  const int b  = bn >> 10;
  const f16 v = (f16)(zr[(size_t)bn * 128 + d] * state[idx]);
  X[(size_t)bn * (2 * CP) + Din + d] = v;
  Xt[((size_t)b * CP + Din + d) * NN + n] = v;
}

// GRU combine + HiPPO-LegS memory update
__global__ void __launch_bounds__(256) k_state_update(int layer, int t,
    const float* __restrict__ zr, const float* __restrict__ hc,
    float* __restrict__ state, float* __restrict__ cbuf,
    float* __restrict__ hs0, float* __restrict__ outMain, float* __restrict__ outHid) {
  int idx = blockIdx.x * blockDim.x + threadIdx.x;
  if (idx >= BB * NN * DO) return;
  const int d  = idx & (DO - 1);
  const int bn = idx >> 6;
  const int n  = bn & (NN - 1);
  const int b  = bn >> 10;
  const float r = zr[(size_t)bn * 128 + DO + d];
  const float h = r * state[idx] + (1.f - r) * hc[idx];
  state[idx] = h;
  if (layer == 0) hs0[(((size_t)t * BB + b) * NN + n) * DO + d] = h;     // next-layer input [T,B,N,D]
  else            outMain[(((size_t)b * TT + t) * NN + n) * DO + d] = h; // output [B,T,N,D]
  if (t == TT - 1) outHid[(((size_t)layer * BB + b) * NN + n) * DO + d] = h;
  const float dt = 1.f / (float)(t + 1);
  float c[ORD], rr[ORD];
  float* cp = cbuf + (size_t)idx * ORD;
#pragma unroll
  for (int o = 0; o < ORD; ++o) { c[o] = cp[o]; rr[o] = sqrtf(2.f * o + 1.f); }
#pragma unroll
  for (int p = 0; p < ORD; ++p) {
    float ac = (p + 1.f) * c[p];                       // diag(n+1)
#pragma unroll
    for (int o = 0; o < ORD; ++o)
      if (o < p) ac += rr[p] * rr[o] * c[o];           // tril(r r^T, -1)
    cp[p] = c[p] - dt * ac + dt * h * rr[p];           // Bv = r
  }
}

// ---------------- host orchestration ----------------
extern "C" void kernel_launch(void* const* d_in, const int* in_sizes, int n_in,
                              void* d_out, int out_size, void* d_ws, size_t ws_size,
                              hipStream_t stream) {
  (void)in_sizes; (void)n_in; (void)out_size; (void)ws_size;
  const float* x     = (const float*)d_in[0];
  const float* ist   = (const float*)d_in[1];
  const float* ic    = (const float*)d_in[2];
  const float* E     = (const float*)d_in[3];
  const float* gp[2] = {(const float*)d_in[4],  (const float*)d_in[9]};
  const float* gb[2] = {(const float*)d_in[5],  (const float*)d_in[10]};
  const float* up[2] = {(const float*)d_in[6],  (const float*)d_in[11]};
  const float* ub[2] = {(const float*)d_in[7],  (const float*)d_in[12]};
  const float* rd[2] = {(const float*)d_in[8],  (const float*)d_in[13]};
  float* outMain = (float*)d_out;
  float* outHid  = outMain + (size_t)BB * TT * NN * DO;

  char* w = (char*)d_ws;
  auto alloc = [&](size_t bytes) -> char* {
    char* p = w;
    w += (bytes + 255) & ~(size_t)255;
    return p;
  };
  const int CPa[2]  = {160, 192};   // Cin padded to mult of 32 (130->160, 192->192)
  const int Cina[2] = {130, 192};
  const int Dina[2] = {2, 64};

  f16* S16 = (f16*)alloc((size_t)NN * NN * 2);
  f16* WgE[2]; f16* WuE[2]; float* bgB[2]; float* buB[2];
  for (int l = 0; l < 2; ++l) {
    WgE[l] = (f16*)alloc((size_t)EDIM * 2 * CPa[l] * 128 * 2);
    WuE[l] = (f16*)alloc((size_t)EDIM * 2 * CPa[l] * 64 * 2);
    bgB[l] = (float*)alloc((size_t)NN * 128 * 4);
    buB[l] = (float*)alloc((size_t)NN * 64 * 4);
  }
  float* state = (float*)alloc((size_t)BB * NN * DO * 4);
  float* cbuf  = (float*)alloc((size_t)BB * NN * DO * ORD * 4);
  float* hs0   = (float*)alloc((size_t)TT * BB * NN * DO * 4);
  f16*   Xflat = (f16*)alloc((size_t)BB * NN * 2 * 192 * 2);
  f16*   Xt    = (f16*)alloc((size_t)BB * 192 * NN * 2);
  float* zr    = (float*)alloc((size_t)BB * NN * 128 * 4);
  float* hc    = (float*)alloc((size_t)BB * NN * DO * 4);

  // ---- phase 0: adjacency + weight/bias preparation (recomputed every call) ----
  k_compute_S<<<NN, 256, 0, stream>>>(E, S16);
  for (int l = 0; l < 2; ++l) {
    const int CP = CPa[l], Cin = Cina[l];
    int tot = EDIM * 2 * CP * 128;
    k_pack_pool_t<<<(tot + 255) / 256, 256, 0, stream>>>(gp[l], WgE[l], Cin, CP, 128);
    tot = EDIM * 2 * CP * 64;
    k_pack_pool_t<<<(tot + 255) / 256, 256, 0, stream>>>(up[l], WuE[l], Cin, CP, 64);
    k_mix_bias<<<(NN * 128 + 255) / 256, 256, 0, stream>>>(E, gb[l], bgB[l], 128);
    k_mix_bias<<<(NN * 64 + 255) / 256, 256, 0, stream>>>(E, ub[l], buB[l], 64);
  }

  const int PT = BB * NN * DO;
  for (int l = 0; l < 2; ++l) {
    hipMemcpyAsync(state, ist + (size_t)l * BB * NN * DO,
                   (size_t)BB * NN * DO * 4, hipMemcpyDeviceToDevice, stream);
    hipMemcpyAsync(cbuf, ic + (size_t)l * BB * NN * DO * ORD,
                   (size_t)BB * NN * DO * ORD * 4, hipMemcpyDeviceToDevice, stream);
    const int CP = CPa[l], Din = Dina[l];
    const int KC = 2 * CP;
    const int ntiles2 = CP / 32;            // 32-col blocks (5 or 6)
    const float* xsrc = (l == 0) ? x : hs0;
    for (int t = 0; t < TT; ++t) {
      k_build_xcat<<<(PT + 255) / 256, 256, 0, stream>>>(l, t, xsrc, state, cbuf, rd[l], Xflat, Xt, CP, Din);
      k_spmm<<<dim3(NN / 32, (ntiles2 + 3) / 4, BB), 128, 0, stream>>>(S16, Xt, Xflat, CP, ntiles2);
      k_gemm_gate<<<dim3(BB * NN / 32, 1), 128, 0, stream>>>(Xflat, KC, WgE[l], bgB[l], E, zr);
      k_patch_xu<<<(PT + 255) / 256, 256, 0, stream>>>(zr, state, Xflat, Xt, CP, Din);
      k_spmm<<<dim3(NN / 32, (ntiles2 + 3) / 4, BB), 128, 0, stream>>>(S16, Xt, Xflat, CP, ntiles2);
      k_gemm_update<<<dim3(BB * NN / 32, 1), 128, 0, stream>>>(Xflat, KC, WuE[l], buB[l], E, hc);
      k_state_update<<<(PT + 255) / 256, 256, 0, stream>>>(l, t, zr, hc, state, cbuf, hs0, outMain, outHid);
    }
  }
}
